// QuanTexSRNet_88871463289264
// MI455X (gfx1250) — compile-verified, matrix operand order checked by reading, not run on hardware
//
#include <hip/hip_runtime.h>
#include <hip/hip_bf16.h>

typedef __attribute__((ext_vector_type(16))) _Float16 v16h;
typedef __attribute__((ext_vector_type(8)))  _Float16 v8h;
typedef __attribute__((ext_vector_type(8)))  float    v8f;

#define N_POS    32768   // B*H*W = 32*32*32
#define C_DIM    512
#define N_CODE   1024
#define HW       1024
#define POS_TILE 128
#define LDS_STRIDE 520   // 512 + 8 f16 pad -> conflict-minimal ds_load_b128

// ---------- kernel 1: codebook f32 -> f16, plus ||e||^2 ----------
__global__ void prep_codebook(const float* __restrict__ cb,
                              _Float16* __restrict__ cb16,
                              float* __restrict__ e2) {
  __shared__ float red[256];
  const int r = blockIdx.x;
  const int t = threadIdx.x;
  float acc = 0.f;
  for (int i = t; i < C_DIM; i += 256) {
    float v = cb[(size_t)r * C_DIM + i];
    cb16[(size_t)r * C_DIM + i] = (_Float16)v;
    acc += v * v;
  }
  red[t] = acc;
  __syncthreads();
  for (int s = 128; s > 0; s >>= 1) {
    if (t < s) red[t] += red[t + s];
    __syncthreads();
  }
  if (t == 0) e2[r] = red[0];
}

// ---------- kernel 2: z (B,C,HW) f32 -> zf16 (pos, C) f16 ----------
__global__ void convert_z(const float* __restrict__ z,
                          _Float16* __restrict__ zf16) {
  const int pos = blockIdx.x * 256 + threadIdx.x;
  const int b = pos >> 10;
  const int hw = pos & 1023;
  const float* zb = z + (size_t)b * C_DIM * HW + hw;
  _Float16* dst = zf16 + (size_t)pos * C_DIM;
  for (int c = 0; c < C_DIM; c += 8) {
    v8h v;
#pragma unroll
    for (int j = 0; j < 8; ++j)
      v[j] = (_Float16)zb[(size_t)(c + j) * HW];
    *(v8h*)(dst + c) = v;
  }
}

// ---------- kernel 3: WMMA distance GEMM + running argmin ----------
// D[m, n] = ||e_m||^2 - 2 * (codebook @ z^T)[m, n]   (||z||^2 constant per n)
// Each wave: 16 positions x 32 codes per chunk (2 A tiles share one B load).
__global__ void argmin_kernel(const _Float16* __restrict__ zf16,
                              const _Float16* __restrict__ cb16,
                              const float* __restrict__ e2,
                              int* __restrict__ out_idx,
                              float* __restrict__ out_idx_f) {
  extern __shared__ _Float16 zsh[];   // POS_TILE x LDS_STRIDE
  const int t    = threadIdx.x;
  const int wv   = t >> 5;
  const int lane = t & 31;
  const int nl   = lane & 15;
  const int half = lane >> 4;
  const int wg_base = blockIdx.x * POS_TILE;

  // stage B tile (128 pos x 512 K, f16) into LDS: 2 threads per row
  {
    const int row = t >> 1;
    const int seg = (t & 1) * 256;
    const v8h* src = (const v8h*)(zf16 + (size_t)(wg_base + row) * C_DIM + seg);
    _Float16* drow = zsh + row * LDS_STRIDE + seg;
#pragma unroll
    for (int j = 0; j < 32; ++j)
      *(v8h*)(drow + j * 8) = src[j];
  }
  __syncthreads();

  const _Float16* brow = zsh + (wv * 16 + nl) * LDS_STRIDE + half * 8;
  float best = 3.4e38f;
  int bestm = 0;

  for (int m0 = 0; m0 < N_CODE; m0 += 32) {
    const _Float16* arow0 = cb16 + (size_t)(m0 + nl) * C_DIM + half * 8;
    const _Float16* arow1 = arow0 + 16 * C_DIM;
    if (m0 + 32 < N_CODE)
      __builtin_prefetch(arow0 + 32 * C_DIM, 0, 1);   // global_prefetch next A block
    v8f acc0 = {};
    v8f acc1 = {};
#pragma unroll 4
    for (int kc = 0; kc < C_DIM; kc += 32) {
      v8h a00 = *(const v8h*)(arow0 + kc);
      v8h a01 = *(const v8h*)(arow0 + kc + 16);
      v8h a10 = *(const v8h*)(arow1 + kc);
      v8h a11 = *(const v8h*)(arow1 + kc + 16);
      v8h b0  = *(const v8h*)(brow + kc);
      v8h b1  = *(const v8h*)(brow + kc + 16);
      v16h A0 = __builtin_shufflevector(a00, a01, 0,1,2,3,4,5,6,7,8,9,10,11,12,13,14,15);
      v16h A1 = __builtin_shufflevector(a10, a11, 0,1,2,3,4,5,6,7,8,9,10,11,12,13,14,15);
      v16h Bv = __builtin_shufflevector(b0,  b1,  0,1,2,3,4,5,6,7,8,9,10,11,12,13,14,15);
      acc0 = __builtin_amdgcn_wmma_f32_16x16x32_f16(
          false, A0, false, Bv, (short)0, acc0, false, false);
      acc1 = __builtin_amdgcn_wmma_f32_16x16x32_f16(
          false, A1, false, Bv, (short)0, acc1, false, false);
    }
    const float* e2p0 = e2 + m0 + half * 8;
    const float* e2p1 = e2p0 + 16;
#pragma unroll
    for (int j = 0; j < 8; ++j) {
      float d = e2p0[j] - 2.0f * acc0[j];
      int m = m0 + half * 8 + j;
      if (d < best) { best = d; bestm = m; }
    }
#pragma unroll
    for (int j = 0; j < 8; ++j) {
      float d = e2p1[j] - 2.0f * acc1[j];
      int m = m0 + 16 + half * 8 + j;
      if (d < best) { best = d; bestm = m; }
    }
  }
  // lanes n and n+16 hold same position, disjoint code halves -> merge
  float ob = __shfl_xor(best, 16, 32);
  int   om = __shfl_xor(bestm, 16, 32);
  if (ob < best || (ob == best && om < bestm)) { best = ob; bestm = om; }

  if (half == 0) {
    const int pos = wg_base + wv * 16 + nl;
    out_idx[pos]   = bestm;
    out_idx_f[pos] = (float)bestm;
  }
}

// ---------- kernel 4: gather z_q (exact f32) + per-(b,c) sq-err partials ----------
__global__ void gather_loss(const float* __restrict__ z,
                            const float* __restrict__ cb,
                            const int* __restrict__ idx,
                            float* __restrict__ zq,
                            float* __restrict__ partials) {
  __shared__ float red[256];
  const int blk = blockIdx.x;     // b*512 + c
  const int b = blk >> 9;
  const int t = threadIdx.x;
  const size_t base = (size_t)blk * HW;   // ((b*512)+c)*1024
  const int c = blk & 511;
  float acc = 0.f;
  for (int hw = t; hw < HW; hw += 256) {
    int k = idx[b * HW + hw];
    float cv = cb[(size_t)k * C_DIM + c];
    float zv = z[base + hw];
    zq[base + hw] = cv;
    float d = cv - zv;
    acc += d * d;
  }
  red[t] = acc;
  __syncthreads();
  for (int s = 128; s > 0; s >>= 1) {
    if (t < s) red[t] += red[t + s];
    __syncthreads();
  }
  if (t == 0) partials[blk] = red[0];
}

// ---------- kernel 5: deterministic final reduction ----------
__global__ void finalize(const float* __restrict__ partials,
                         float* __restrict__ out_loss) {
  __shared__ float red[256];
  const int t = threadIdx.x;
  float acc = 0.f;
  for (int j = 0; j < 64; ++j) acc += partials[t * 64 + j];
  red[t] = acc;
  __syncthreads();
  for (int s = 128; s > 0; s >>= 1) {
    if (t < s) red[t] += red[t + s];
    __syncthreads();
  }
  if (t == 0) {
    float mse = red[0] / 16777216.0f;   // mean over N*C
    out_loss[0] = 1.25f * mse;          // q_latent + 0.25 * e_latent
    out_loss[1] = 0.0f;                 // cls_loss
  }
}

extern "C" void kernel_launch(void* const* d_in, const int* in_sizes, int n_in,
                              void* d_out, int out_size, void* d_ws, size_t ws_size,
                              hipStream_t stream) {
  const float* z  = (const float*)d_in[0];   // (32,512,32,32)
  const float* cb = (const float*)d_in[1];   // (1024,512)
  float* out = (float*)d_out;

  // workspace layout
  char* ws = (char*)d_ws;
  _Float16* cb16     = (_Float16*)ws;                                   // 1 MiB
  float*    e2       = (float*)(ws + (1u << 20));                       // 4 KiB
  _Float16* zf16     = (_Float16*)(ws + (1u << 20) + 4096);             // 32 MiB
  char*     after_z  = ws + (1u << 20) + 4096 + (size_t)N_POS * C_DIM * 2;
  int*      idx      = (int*)after_z;                                   // 128 KiB
  float*    partials = (float*)(after_z + (size_t)N_POS * 4);           // 64 KiB

  // output layout: z_q (16777216) | codebook_loss | cls_loss | indices (32768)
  float* zq    = out;
  float* loss  = out + 16777216;
  float* idx_f = out + 16777218;

  prep_codebook<<<N_CODE, 256, 0, stream>>>(cb, cb16, e2);
  convert_z<<<N_POS / 256, 256, 0, stream>>>(z, zf16);

  const size_t shmem = (size_t)POS_TILE * LDS_STRIDE * sizeof(_Float16); // 133,120 B (<320 KB WGP LDS)
  argmin_kernel<<<N_POS / POS_TILE, 256, shmem, stream>>>(zf16, cb16, e2, idx, idx_f);

  gather_loss<<<32 * 512, 256, 0, stream>>>(z, cb, idx, zq, partials);
  finalize<<<1, 256, 0, stream>>>(partials, loss);
}
